// PSDNet_81372450390611
// MI455X (gfx1250) — matrix operand order repaired and checked by reading, not compiled, not run on hardware
//
#include <hip/hip_runtime.h>
#include <cmath>

#define LEAK 0.01f
#define NSWEEP 10

typedef __attribute__((ext_vector_type(16))) _Float16 v16h;
typedef __attribute__((ext_vector_type(8)))  float    v8f;

// ---------------- workspace layout (bytes) ----------------
// wf1 @ 0      : 16 tiles * 512 halves = 16384 B   (W1: K=64,  N=128, KT=2, NT=8)
// wf2 @ 16384  : 32 tiles * 512 halves = 32768 B   (W2: K=128, N=128, KT=4, NT=8)
// wf3 @ 49152  : 36 tiles * 512 halves = 36864 B   (W3: K=128, N=144(pad), KT=4, NT=9)
// b3p @ 86016  : 144 floats = 576 B

// ---------------- weight repack into WMMA B-fragment layout ----------------
// B fragment (16x16x32 f16): lane l holds column N = l&15; K = 32*kt + 16*(l>=16) + e.
// Stored as [tile = nt*KT + kt][lane][e] -> 32B contiguous per lane.
__global__ __launch_bounds__(256) void psd_prep(
    const float* __restrict__ W1, const float* __restrict__ W2,
    const float* __restrict__ W3, const float* __restrict__ b3,
    unsigned char* __restrict__ ws)
{
  int t = blockIdx.x * 256 + threadIdx.x;
  float* b3p = (float*)(ws + 86016);
  if (t < 144) b3p[t] = (t < 136) ? b3[t] : 0.0f;
  if (t >= 43008) return;

  const float* W; _Float16* wf; int Nr, KT, local;
  if (t < 8192)       { W = W1; wf = (_Float16*)(ws);         Nr = 128; KT = 2; local = t; }
  else if (t < 24576) { W = W2; wf = (_Float16*)(ws + 16384); Nr = 128; KT = 4; local = t - 8192; }
  else                { W = W3; wf = (_Float16*)(ws + 49152); Nr = 136; KT = 4; local = t - 24576; }

  int tile = local >> 9;
  int rem  = local & 511;
  int lane = rem >> 4;
  int e    = rem & 15;
  int kt   = tile % KT;
  int nt   = tile / KT;
  int K    = kt * 32 + ((lane & 16) ? 16 : 0) + e;
  int N    = nt * 16 + (lane & 15);
  float v  = (N < Nr) ? W[K * Nr + N] : 0.0f;
  wf[local] = (_Float16)v;
}

// A fragment (16x16x32 f16, 16xK tile): lane l holds row M = l&15;
// element e -> K = 32*kt + 16*(l>=16) + e.
__device__ inline v16h afrag_f32(const float* base, int ldK, int kt, int lane) {
  int M = lane & 15;
  int khi = (lane & 16) ? 16 : 0;
  v16h a;
#pragma unroll
  for (int e = 0; e < 16; ++e) {
    int K = kt * 32 + khi + e;
    a[e] = (_Float16)base[M * ldK + K];
  }
  return a;
}
__device__ inline v16h afrag_f16(const _Float16* base, int ldK, int kt, int lane) {
  int M = lane & 15;
  int khi = (lane & 16) ? 16 : 0;
  v16h a;
#pragma unroll
  for (int e = 0; e < 16; ++e) {
    int K = kt * 32 + khi + e;
    a[e] = base[M * ldK + K];
  }
  return a;
}

// ---------------- fused MLP: ldata = tanh(L3(lrelu(L2(lrelu(L1(x)))))) ----------------
// 8 waves / block; each wave owns a 16-row batch tile. Writes 136 floats per
// element into the front of its 256-float d_out slot (staging for psd_eigen).
__global__ __launch_bounds__(256) void psd_mlp(
    const float* __restrict__ x, const float* __restrict__ b1,
    const float* __restrict__ b2, const unsigned char* __restrict__ ws,
    float* __restrict__ out)
{
  __shared__ __align__(16) unsigned char smem[8 * 4096];
  const _Float16* wf1 = (const _Float16*)(ws);
  const _Float16* wf2 = (const _Float16*)(ws + 16384);
  const _Float16* wf3 = (const _Float16*)(ws + 49152);
  const float*    b3p = (const float*)(ws + 86016);

  int lane = threadIdx.x & 31;
  int wave = threadIdx.x >> 5;
  int row0 = (blockIdx.x * 8 + wave) * 16;
  unsigned char* my = smem + wave * 4096;
  float*    xs = (float*)my;      // 16x64 f32 staging (4KB)
  _Float16* hs = (_Float16*)my;   // 16x128 f16 activations (4KB, reused)

  // stage x tile (coalesced float4)
  {
    const float4* src = (const float4*)(x + (size_t)row0 * 64);
    float4* dst = (float4*)xs;
#pragma unroll
    for (int i = 0; i < 8; ++i) dst[lane + i * 32] = src[lane + i * 32];
  }
  __syncthreads();

  int colb = lane & 15;
  int mhi  = (lane & 16) ? 8 : 0;
  v8f acc[8];

  // ---- layer 1: 16x64 @ 64x128 ----
  v16h a1[2];
#pragma unroll
  for (int kt = 0; kt < 2; ++kt) a1[kt] = afrag_f32(xs, 64, kt, lane);
#pragma unroll
  for (int nt = 0; nt < 8; ++nt) {
    v8f c = {};
#pragma unroll
    for (int kt = 0; kt < 2; ++kt) {
      v16h bf = *(const v16h*)(wf1 + (nt * 2 + kt) * 512 + lane * 16);
      c = __builtin_amdgcn_wmma_f32_16x16x32_f16(false, a1[kt], false, bf,
                                                 (short)0, c, false, false);
    }
    acc[nt] = c;
  }
  __syncthreads();
#pragma unroll
  for (int nt = 0; nt < 8; ++nt) {
    int col = nt * 16 + colb;
    float bias = b1[col];
#pragma unroll
    for (int e = 0; e < 8; ++e) {
      float v = acc[nt][e] + bias;
      v = (v > 0.0f) ? v : LEAK * v;
      hs[(e + mhi) * 128 + col] = (_Float16)v;   // C/D layout: M = e + 8*(l>=16), N = l&15
    }
  }
  __syncthreads();

  // ---- layer 2: 16x128 @ 128x128 ----
  v16h a2[4];
#pragma unroll
  for (int kt = 0; kt < 4; ++kt) a2[kt] = afrag_f16(hs, 128, kt, lane);
#pragma unroll
  for (int nt = 0; nt < 8; ++nt) {
    v8f c = {};
#pragma unroll
    for (int kt = 0; kt < 4; ++kt) {
      v16h bf = *(const v16h*)(wf2 + (nt * 4 + kt) * 512 + lane * 16);
      c = __builtin_amdgcn_wmma_f32_16x16x32_f16(false, a2[kt], false, bf,
                                                 (short)0, c, false, false);
    }
    acc[nt] = c;
  }
  __syncthreads();
#pragma unroll
  for (int nt = 0; nt < 8; ++nt) {
    int col = nt * 16 + colb;
    float bias = b2[col];
#pragma unroll
    for (int e = 0; e < 8; ++e) {
      float v = acc[nt][e] + bias;
      v = (v > 0.0f) ? v : LEAK * v;
      hs[(e + mhi) * 128 + col] = (_Float16)v;
    }
  }
  __syncthreads();

  // ---- layer 3: 16x128 @ 128x144(pad), tanh, store first 136 cols ----
  v16h a3[4];
#pragma unroll
  for (int kt = 0; kt < 4; ++kt) a3[kt] = afrag_f16(hs, 128, kt, lane);
#pragma unroll
  for (int nt = 0; nt < 9; ++nt) {
    v8f c = {};
#pragma unroll
    for (int kt = 0; kt < 4; ++kt) {
      v16h bf = *(const v16h*)(wf3 + (nt * 4 + kt) * 512 + lane * 16);
      c = __builtin_amdgcn_wmma_f32_16x16x32_f16(false, a3[kt], false, bf,
                                                 (short)0, c, false, false);
    }
    int col = nt * 16 + colb;
    if (col < 136) {
      float bias = b3p[col];
#pragma unroll
      for (int e = 0; e < 8; ++e) {
        float v = tanhf(c[e] + bias);
        out[(size_t)(row0 + e + mhi) * 256 + col] = v;
      }
    }
  }
}

// ---------------- parallel Jacobi eigensolver + reconstruction ----------------
__device__ inline int mod15(int x) { x %= 15; return (x < 0) ? x + 15 : x; }
// round-robin tournament pairing: 15 rounds x 8 disjoint pairs covering all 120 (p,q)
__device__ inline void pair_pq(int r, int k, int* pp, int* qq) {
  int p, q;
  if (k == 0) { p = 0; q = mod15(0 - r) + 1; }
  else        { p = mod15(k - r) + 1; q = mod15(15 - k - r) + 1; }
  if (p > q) { int t = p; p = q; q = t; }
  *pp = p; *qq = q;
}

__global__ __launch_bounds__(256) void psd_eigen(float* __restrict__ out)
{
  __shared__ float Qs[16][16][17];
  __shared__ float Vs[16][16][17];
  __shared__ float Cs[16][8];
  __shared__ float Ss[16][8];
  __shared__ float Ev[16][16];

  int lane = threadIdx.x & 31;
  int wave = threadIdx.x >> 5;
  int half = lane >> 4;
  int j    = lane & 15;            // this lane owns column j
  int m    = wave * 2 + half;      // matrix slot in block (0..15)
  size_t b = (size_t)blockIdx.x * 16 + m;
  float* base = out + b * 256;     // first 136 floats hold staged ldata

  // build Q = L + L^T from tril-scattered ldata
#pragma unroll
  for (int i = 0; i < 16; ++i) {
    int r = (i > j) ? i : j;
    int c = (i > j) ? j : i;
    float v = base[r * (r + 1) / 2 + c];
    if (i == j) v *= 2.0f;
    Qs[m][i][j] = v;
    Vs[m][i][j] = (i == j) ? 1.0f : 0.0f;
  }
  __syncthreads();

  for (int sw = 0; sw < NSWEEP; ++sw) {
    for (int r = 0; r < 15; ++r) {
      // lanes 0..7 compute the 8 rotation angles
      if (j < 8) {
        int p, q; pair_pq(r, j, &p, &q);
        float app = Qs[m][p][p], aqq = Qs[m][q][q], apq = Qs[m][p][q];
        float cc = 1.0f, ss = 0.0f;
        if (fabsf(apq) > 1e-20f) {
          float tau = (aqq - app) / (2.0f * apq);
          float tt  = copysignf(1.0f, tau) / (fabsf(tau) + sqrtf(1.0f + tau * tau));
          cc = 1.0f / sqrtf(1.0f + tt * tt);
          ss = tt * cc;
        }
        Cs[m][j] = cc; Ss[m][j] = ss;
      }
      __syncthreads();
      // column rotations (A <- A*J, V <- V*J): lane handles row i = j
#pragma unroll
      for (int k = 0; k < 8; ++k) {
        int p, q; pair_pq(r, k, &p, &q);
        float cc = Cs[m][k], ss = Ss[m][k];
        float qp = Qs[m][j][p], qq2 = Qs[m][j][q];
        Qs[m][j][p] = cc * qp - ss * qq2;
        Qs[m][j][q] = ss * qp + cc * qq2;
        float vp = Vs[m][j][p], vq = Vs[m][j][q];
        Vs[m][j][p] = cc * vp - ss * vq;
        Vs[m][j][q] = ss * vp + cc * vq;
      }
      __syncthreads();
      // row rotations (A <- J^T*A): lane handles column i = j
#pragma unroll
      for (int k = 0; k < 8; ++k) {
        int p, q; pair_pq(r, k, &p, &q);
        float cc = Cs[m][k], ss = Ss[m][k];
        float qp = Qs[m][p][j], qq2 = Qs[m][q][j];
        Qs[m][p][j] = cc * qp - ss * qq2;
        Qs[m][q][j] = ss * qp + cc * qq2;
      }
      __syncthreads();
    }
  }

  // ev = sqrt(softmax(exp(2w)))  (order-invariant -> no eigenvalue sort needed)
  float w  = Qs[m][j][j];
  float e  = __expf(2.0f * w);
  float mx = e;
#pragma unroll
  for (int d = 1; d < 16; d <<= 1) mx = fmaxf(mx, __shfl_xor(mx, d, 32));
  float pe  = __expf(e - mx);
  float sum = pe;
#pragma unroll
  for (int d = 1; d < 16; d <<= 1) sum += __shfl_xor(sum, d, 32);
  Ev[m][j] = sqrtf(pe / sum);
  __syncthreads();

  // O = V diag(ev) V^T ; lane j emits column j (overwrites staging)
  float coef[16];
#pragma unroll
  for (int jj = 0; jj < 16; ++jj) coef[jj] = Ev[m][jj] * Vs[m][j][jj];
#pragma unroll
  for (int i = 0; i < 16; ++i) {
    float o = 0.0f;
#pragma unroll
    for (int jj = 0; jj < 16; ++jj) o += coef[jj] * Vs[m][i][jj];
    base[i * 16 + j] = o;
  }
}

extern "C" void kernel_launch(void* const* d_in, const int* in_sizes, int n_in,
                              void* d_out, int out_size, void* d_ws, size_t ws_size,
                              hipStream_t stream) {
  const float* x  = (const float*)d_in[0];
  const float* W1 = (const float*)d_in[1];
  const float* b1 = (const float*)d_in[2];
  const float* W2 = (const float*)d_in[3];
  const float* b2 = (const float*)d_in[4];
  const float* W3 = (const float*)d_in[5];
  const float* b3 = (const float*)d_in[6];
  float* out = (float*)d_out;
  unsigned char* ws = (unsigned char*)d_ws;

  psd_prep <<<168,  256, 0, stream>>>(W1, W2, W3, b3, ws);
  psd_mlp  <<<512,  256, 0, stream>>>(x, b1, b2, ws, out);   // 65536 / 128 rows
  psd_eigen<<<4096, 256, 0, stream>>>(out);                  // 65536 / 16 matrices
}
